// GraphAttentionLayer_37795712205407
// MI455X (gfx1250) — matrix operand order repaired and checked by reading, not compiled, not run on hardware
//
#include <hip/hip_runtime.h>
#include <hip/hip_bf16.h>
#include <stdint.h>

#define B_    64
#define N_    512
#define FIN_  128
#define FOUT_ 64
#define ALPHA 0.2f
#define NEGINF (-10000000000000.0f)

typedef __attribute__((ext_vector_type(16))) __bf16 v16bf;
typedef __attribute__((ext_vector_type(8)))  float  v8f;

// float -> bf16 bits, round-half-up (cheap: add + shift)
static __device__ __forceinline__ uint16_t bfbits(float f) {
  return (uint16_t)((__float_as_uint(f) + 0x8000u) >> 16);
}
// pack two f32 -> dword of two bf16 (x in [15:0], y in [31:16])
static __device__ __forceinline__ uint32_t pkbf(float x, float y) {
#if __has_builtin(__builtin_amdgcn_cvt_pk_bf16_f32)
  auto r = __builtin_amdgcn_cvt_pk_bf16_f32(x, y);        // v_cvt_pk_bf16_f32
  uint32_t u; __builtin_memcpy(&u, &r, 4); return u;
#elif __has_builtin(__builtin_amdgcn_perm)
  uint32_t ux = __float_as_uint(x) + 0x8000u;             // round-half-up
  uint32_t uy = __float_as_uint(y) + 0x8000u;
  return __builtin_amdgcn_perm(uy, ux, 0x07060302u);      // one v_perm_b32
#else
  uint32_t ux = __float_as_uint(x), uy = __float_as_uint(y);
  return ((ux + 0x8000u) >> 16) | (((uy + 0x8000u) >> 16) << 16);
#endif
}

union AFrag { v16bf v; uint32_t d[8]; };
union BFrag { v16bf v; uint4 q[2]; };

// LDS row strides (in u16) for transposed B panels; chosen so the per-lane
// 32B fragment runs start 4 banks apart mod 64 -> conflict-free ds_load_b128.
#define SP_STRIDE 40    // 80B  = 20 banks step
#define HT_STRIDE 40
#define WT_STRIDE 136   // 272B = 68 banks step (=4 mod 64)

// ---------------------------------------------------------------- kernel 1
// Convert W (128x64) and sparse (512x512) to bf16 once (dword-packed).
__global__ __launch_bounds__(256) void prep_cvt(
    const float* __restrict__ W, const float* __restrict__ sparse,
    uint32_t* __restrict__ Wb, uint32_t* __restrict__ Sb) {
  int i = blockIdx.x * 256 + threadIdx.x;      // dword index
  if (i < FIN_ * FOUT_ / 2) {
    float2 p = ((const float2*)W)[i];
    Wb[i] = pkbf(p.x, p.y);
  }
  if (i < N_ * N_ / 2) {
    float2 p = ((const float2*)sparse)[i];
    Sb[i] = pkbf(p.x, p.y);
  }
}

// ---------------------------------------------------------------- kernel 2
// h = inp @ W   (32768x128 @ 128x64), bf16 WMMA, f32 accumulate.
// Block: 64 rows x 64 cols, 8 waves: wave w -> rows (w%4)*16, cols (w/4)*32.
__global__ __launch_bounds__(256) void gemm_h(
    const float* __restrict__ inp, const uint16_t* __restrict__ Wb,
    float* __restrict__ h32, uint16_t* __restrict__ h16) {
  __shared__ uint16_t wTt[FOUT_ * WT_STRIDE];   // W transposed [n][k], padded
  const int t = threadIdx.x;
  for (int e = t; e < FIN_ * FOUT_; e += 256) { // stage + transpose once
    int k = e >> 6, n = e & 63;
    wTt[n * WT_STRIDE + k] = Wb[e];
  }
  __syncthreads();

  const int r0 = blockIdx.x * 64;
  const int w = t >> 5, l = t & 31;
  const int mBase = (w & 3) * 16, nBase = (w >> 2) * 32;
  const int m = l & 15, lh = (l >> 4) & 1;

  v8f acc0 = {}, acc1 = {};
  const float* arowBase = inp + (size_t)(r0 + mBase + m) * FIN_;
  for (int kk = 0; kk < FIN_; kk += 32) {
    AFrag a;
    {
      const float4* p0 = (const float4*)(arowBase + kk + lh * 16);
      const float4* p1 = (const float4*)(arowBase + kk + lh * 16 + 8);
      float4 x0 = p0[0], x1 = p0[1], x2 = p1[0], x3 = p1[1];
      a.d[0] = pkbf(x0.x, x0.y); a.d[1] = pkbf(x0.z, x0.w);
      a.d[2] = pkbf(x1.x, x1.y); a.d[3] = pkbf(x1.z, x1.w);
      a.d[4] = pkbf(x2.x, x2.y); a.d[5] = pkbf(x2.z, x2.w);
      a.d[6] = pkbf(x3.x, x3.y); a.d[7] = pkbf(x3.z, x3.w);
    }
    BFrag b0, b1;
    {
      const uint4* q0 = (const uint4*)(wTt + (nBase + m) * WT_STRIDE + kk + lh * 16);
      const uint4* q1 = (const uint4*)(wTt + (nBase + 16 + m) * WT_STRIDE + kk + lh * 16);
      b0.q[0] = q0[0]; b0.q[1] = q0[1];          // 2x ds_load_b128 each
      b1.q[0] = q1[0]; b1.q[1] = q1[1];
    }
    acc0 = __builtin_amdgcn_wmma_f32_16x16x32_bf16(false, a.v, false, b0.v, (short)0, acc0, false, false);
    acc1 = __builtin_amdgcn_wmma_f32_16x16x32_bf16(false, a.v, false, b1.v, (short)0, acc1, false, false);
  }
#pragma unroll
  for (int r = 0; r < 8; ++r) {
    int mm = mBase + r + lh * 8;
    size_t o0 = (size_t)(r0 + mm) * FOUT_ + nBase + m;
    size_t o1 = o0 + 16;
    h32[o0] = acc0[r];          h32[o1] = acc1[r];
    h16[o0] = bfbits(acc0[r]);  h16[o1] = bfbits(acc1[r]);
  }
}

// ---------------------------------------------------------------- kernel 3
// s1 = h @ a1, s2 = h @ a2 (per flattened row of h).
__global__ __launch_bounds__(256) void scores(
    const float* __restrict__ h32, const float* __restrict__ a,
    float* __restrict__ s1, float* __restrict__ s2) {
  int R = blockIdx.x * 256 + threadIdx.x;
  if (R >= B_ * N_) return;
  const float4* row = (const float4*)(h32 + (size_t)R * FOUT_);
  float x1 = 0.f, x2 = 0.f;
#pragma unroll
  for (int c = 0; c < FOUT_ / 4; ++c) {
    float4 v = row[c];
    const float4 a1 = ((const float4*)a)[c];
    const float4 a2 = ((const float4*)(a + FOUT_))[c];
    x1 += v.x * a1.x + v.y * a1.y + v.z * a1.z + v.w * a1.w;
    x2 += v.x * a2.x + v.y * a2.y + v.z * a2.z + v.w * a2.w;
  }
  s1[R] = x1;
  s2[R] = x2;
}

// ---------------------------------------------------------------- kernel 4
// Fused: masked leaky-relu scores (A built in-registers) @ sparse -> row softmax.
// Block owns rows [i0,i0+16) of batch b, all 512 cols. 8 waves, wave w -> cols [64w,64w+64).
__global__ __launch_bounds__(256) void gemm_att_softmax(
    const int* __restrict__ adj, const uint16_t* __restrict__ Sb,
    const float* __restrict__ s1, const float* __restrict__ s2,
    float* __restrict__ attOut) {
  // 40KB buffer aliased: transposed sparse K-panel (bf16) during GEMM,
  // f32 logits (32KB) for the softmax epilogue.
  __shared__ uint32_t smemBig[N_ * SP_STRIDE / 2];   // 40 KB
  __shared__ int   adjT[16 * 32];                    // 2 KB
  __shared__ float s2col[N_];                        // 2 KB
  __shared__ float s1row[16];
  __shared__ float pred[16 * 16];
  __shared__ float rmax[16], rsum[16];
  uint16_t* spTt   = (uint16_t*)smemBig;             // [n][k] padded, n=0..511,k=0..31
  float*    logits = (float*)smemBig;                // [16][512]

  const int t = threadIdx.x;
  const int b  = blockIdx.x >> 5;
  const int i0 = (blockIdx.x & 31) << 4;

  if (t < 16) s1row[t] = s1[b * N_ + i0 + t];
  for (int c = t; c < N_; c += 256) s2col[c] = s2[b * N_ + c];
  __syncthreads();

  const int w = t >> 5, l = t & 31;
  const int m = l & 15, lh = (l >> 4) & 1;
  const int nB = w * 64;
  const float s1m = s1row[m];
  v8f acc[4] = {v8f{}, v8f{}, v8f{}, v8f{}};

  const int* adjBase = adj + (size_t)b * N_ * N_ + (size_t)i0 * N_;

  for (int kk = 0; kk < N_; kk += 32) {
    __syncthreads();
    // stage adj tile 16x32 (coalesced dwords)
#pragma unroll
    for (int it = 0; it < 2; ++it) {
      int e = t + it * 256;
      adjT[e] = adjBase[(e >> 5) * N_ + kk + (e & 31)];
    }
    // stage sparse panel rows [kk,kk+32) transposed into [n][k] (u16, coalesced in n)
    for (int e = t; e < 32 * N_; e += 256) {
      int k = e >> 9, n = e & (N_ - 1);
      spTt[n * SP_STRIDE + k] = Sb[(kk + k) * N_ + n];
    }
    if (kk + 32 < N_)
      __builtin_prefetch(adjBase + (kk + 32), 0, 0);   // global_prefetch_b8 hint
    __syncthreads();

    // A fragment: masked leaky-relu score, generated from LDS (never hits HBM)
    AFrag a;
    {
      const int2*   aj  = (const int2*)(adjT + m * 32);
      const float2* s2p = (const float2*)(s2col + kk);
#pragma unroll
      for (int v = 0; v < 8; ++v) {
        int k0 = (lh << 4) + (v << 1);
        int2   ad = aj[k0 >> 1];
        float2 sv = s2p[k0 >> 1];
        float sc0 = s1m + sv.x; sc0 = fmaxf(sc0, ALPHA * sc0);  // leaky-relu
        float sc1 = s1m + sv.y; sc1 = fmaxf(sc1, ALPHA * sc1);
        a.d[v] = pkbf(ad.x > 0 ? sc0 : NEGINF, ad.y > 0 ? sc1 : NEGINF);
      }
    }
#pragma unroll
    for (int nt = 0; nt < 4; ++nt) {
      BFrag bb;
      const uint4* q = (const uint4*)(spTt + (nB + nt * 16 + m) * SP_STRIDE + lh * 16);
      bb.q[0] = q[0]; bb.q[1] = q[1];              // 2x ds_load_b128
      acc[nt] = __builtin_amdgcn_wmma_f32_16x16x32_bf16(false, a.v, false, bb.v, (short)0, acc[nt], false, false);
    }
  }
  __syncthreads();
  // spill D tiles into the (now free) buffer as f32 logits
#pragma unroll
  for (int nt = 0; nt < 4; ++nt)
#pragma unroll
    for (int r = 0; r < 8; ++r)
      logits[(r + lh * 8) * N_ + nB + nt * 16 + m] = acc[nt][r];
  __syncthreads();

  // row softmax over 512 cols: 16 threads per row, 32 cols each
  {
    int row = t & 15, cg = t >> 4;
    float mx = -3.4e38f;
    for (int c = cg * 32; c < cg * 32 + 32; ++c) mx = fmaxf(mx, logits[row * N_ + c]);
    pred[row * 16 + cg] = mx;
    __syncthreads();
    if (t < 16) {
      float mm = pred[t * 16];
      for (int g = 1; g < 16; ++g) mm = fmaxf(mm, pred[t * 16 + g]);
      rmax[t] = mm;
    }
    __syncthreads();
    float sm = 0.f;
    for (int c = cg * 32; c < cg * 32 + 32; ++c) {
      float e = __expf(logits[row * N_ + c] - rmax[row]);
      logits[row * N_ + c] = e;
      sm += e;
    }
    pred[row * 16 + cg] = sm;
    __syncthreads();
    if (t < 16) {
      float ss = 0.f;
      for (int g = 0; g < 16; ++g) ss += pred[t * 16 + g];
      rsum[t] = ss;
    }
    __syncthreads();
  }
  float* out = attOut + (size_t)b * N_ * N_ + (size_t)i0 * N_;
  for (int e = t; e < 16 * N_; e += 256) {
    int row = e >> 9;
    out[e] = logits[e] / rsum[row];
  }
}

// ---------------------------------------------------------------- kernel 5
// att = (att + att^T)/2 in place; each (i<j) pair owned by exactly one thread.
__global__ __launch_bounds__(256) void symmetrize(float* __restrict__ att) {
  size_t idx = (size_t)blockIdx.x * 256 + threadIdx.x;
  if (idx >= (size_t)B_ * N_ * N_) return;
  int j = (int)(idx & (N_ - 1));
  int i = (int)((idx >> 9) & (N_ - 1));
  size_t bofs = (idx >> 18) << 18;
  if (i < j) {
    size_t o1 = bofs + ((size_t)i << 9) + j;
    size_t o2 = bofs + ((size_t)j << 9) + i;
    float x = att[o1], y = att[o2];
    float v = 0.5f * (x + y);
    att[o1] = v;
    att[o2] = v;
  }
}

// ---------------------------------------------------------------- kernel 6
// h' = elu(att @ h)  (per batch 512x512 @ 512x64), bf16 WMMA.
// Block: 32 rows x 64 cols; wave w -> rows (w/4)*16, cols (w%4)*16.
__global__ __launch_bounds__(256) void gemm_out(
    const float* __restrict__ att, const uint16_t* __restrict__ h16,
    float* __restrict__ hp) {
  __shared__ uint16_t hTt[FOUT_ * HT_STRIDE];   // h panel transposed [n][k], 5 KB
  const int t = threadIdx.x;
  const int b  = blockIdx.x >> 4;
  const int m0 = (blockIdx.x & 15) << 5;
  const int w = t >> 5, l = t & 31;
  const int mB = (w >> 2) << 4;
  const int nt = (w & 3) << 4;
  const int m = l & 15, lh = (l >> 4) & 1;

  v8f acc = {};
  const float* arow = att + ((size_t)b << 18) + (size_t)(m0 + mB + m) * N_;
  const uint16_t* hSrc = h16 + (size_t)b * N_ * FOUT_;

  for (int kk = 0; kk < N_; kk += 32) {
    __syncthreads();
    for (int e = t; e < 32 * FOUT_; e += 256) {   // stage + transpose (coalesced in n)
      int k = e >> 6, n = e & 63;
      hTt[n * HT_STRIDE + k] = hSrc[(size_t)(kk + k) * FOUT_ + n];
    }
    __syncthreads();

    AFrag a;
    {
      const float4* p0 = (const float4*)(arow + kk + lh * 16);
      const float4* p1 = (const float4*)(arow + kk + lh * 16 + 8);
      float4 x0 = p0[0], x1 = p0[1], x2 = p1[0], x3 = p1[1];
      a.d[0] = pkbf(x0.x, x0.y); a.d[1] = pkbf(x0.z, x0.w);
      a.d[2] = pkbf(x1.x, x1.y); a.d[3] = pkbf(x1.z, x1.w);
      a.d[4] = pkbf(x2.x, x2.y); a.d[5] = pkbf(x2.z, x2.w);
      a.d[6] = pkbf(x3.x, x3.y); a.d[7] = pkbf(x3.z, x3.w);
    }
    BFrag bb;
    {
      const uint4* q = (const uint4*)(hTt + (nt + m) * HT_STRIDE + lh * 16);
      bb.q[0] = q[0]; bb.q[1] = q[1];              // 2x ds_load_b128
    }
    acc = __builtin_amdgcn_wmma_f32_16x16x32_bf16(false, a.v, false, bb.v, (short)0, acc, false, false);
  }
#pragma unroll
  for (int r = 0; r < 8; ++r) {
    float x = acc[r];
    float e = x > 0.f ? x : (__expf(x) - 1.f);   // ELU
    int mm = m0 + mB + r + lh * 8;
    hp[((size_t)b * N_ + mm) * FOUT_ + nt + m] = e;
  }
}

// ----------------------------------------------------------------
extern "C" void kernel_launch(void* const* d_in, const int* in_sizes, int n_in,
                              void* d_out, int out_size, void* d_ws, size_t ws_size,
                              hipStream_t stream) {
  (void)in_sizes; (void)n_in; (void)out_size; (void)ws_size;
  const float* inp    = (const float*)d_in[0];
  const int*   adj    = (const int*)d_in[1];
  const float* W      = (const float*)d_in[2];
  const float* a      = (const float*)d_in[3];
  const float* sparse = (const float*)d_in[4];

  float* hp  = (float*)d_out;                           // h' : B*N*FOUT
  float* att = (float*)d_out + (size_t)B_ * N_ * FOUT_; // att: B*N*N

  char* ws = (char*)d_ws;
  float*    h32 = (float*)ws;     ws += (size_t)B_ * N_ * FOUT_ * 4;
  uint16_t* h16 = (uint16_t*)ws;  ws += (size_t)B_ * N_ * FOUT_ * 2;
  uint16_t* Wb  = (uint16_t*)ws;  ws += (size_t)FIN_ * FOUT_ * 2;
  uint16_t* Sb  = (uint16_t*)ws;  ws += (size_t)N_ * N_ * 2;
  float*    s1  = (float*)ws;     ws += (size_t)B_ * N_ * 4;
  float*    s2  = (float*)ws;     ws += (size_t)B_ * N_ * 4;

  prep_cvt        <<<(N_ * N_ / 2 + 255) / 256, 256, 0, stream>>>(W, sparse, (uint32_t*)Wb, (uint32_t*)Sb);
  gemm_h          <<<B_ * N_ / 64,              256, 0, stream>>>(inp, Wb, h32, h16);
  scores          <<<(B_ * N_ + 255) / 256,     256, 0, stream>>>(h32, a, s1, s2);
  gemm_att_softmax<<<B_ * (N_ / 16),            256, 0, stream>>>(adj, Sb, s1, s2, att);
  symmetrize      <<<(B_ * N_ * N_) / 256,      256, 0, stream>>>(att);
  gemm_out        <<<B_ * (N_ / 32),            256, 0, stream>>>(att, h16, hp);
}
